// Actor_77412490543294
// MI455X (gfx1250) — compile-verified, matrix operand order checked by reading, not run on hardware
//
#include <hip/hip_runtime.h>
#include <hip/hip_bf16.h>

typedef __attribute__((ext_vector_type(16))) _Float16 v16h;
typedef __attribute__((ext_vector_type(8)))  _Float16 v8h;
typedef __attribute__((ext_vector_type(8)))  float    v8f;

#define BATCH   64
#define LSEQ    1000
#define EDIM    128
#define DDIM    256
#define NSTATE  16
#define SDIM    128
#define DTRANK  8
#define KCONV   4
#define DIN     256            // 2*E
#define TWO_D   512            // 2*D
#define MROWS   (BATCH * LSEQ) // 64000
#define NT_PER_WAVE 4          // N-tiles per wave in the big GEMM

// ---------------- Embedding: h_in = [x@W_emb+b_emb || pos_emb], stored f16 ----
__global__ void k_embed(const float* __restrict__ x, const float* __restrict__ W_emb,
                        const float* __restrict__ b_emb, const float* __restrict__ pos_emb,
                        _Float16* __restrict__ hin) {
    int i = blockIdx.x * blockDim.x + threadIdx.x; // over MROWS*DIN
    if (i >= MROWS * DIN) return;
    int row = i >> 8;        // / 256
    int k   = i & 255;
    int t   = row % LSEQ;
    float v;
    if (k < EDIM) {
        v = x[row * 2 + 0] * W_emb[k] + x[row * 2 + 1] * W_emb[EDIM + k] + b_emb[k];
    } else {
        v = pos_emb[t * EDIM + (k - EDIM)];
    }
    hin[i] = (_Float16)v;
}

// ---------------- Pack W_in (f32 [256,512]) into WMMA B-fragment order, f16 ---
// packed[((ntile*8 + kchunk)*32 + lane)*16 + e] = W_in[K][N]
//   K = kchunk*32 + (e&7) + (e>>3)*16 + (lane>>4)*8,  N = ntile*16 + (lane&15)
__global__ void k_packW(const float* __restrict__ W_in, _Float16* __restrict__ wpk) {
    int i = blockIdx.x * blockDim.x + threadIdx.x; // 0 .. 256*512
    if (i >= DIN * TWO_D) return;
    int e      = i & 15;
    int lane   = (i >> 4) & 31;
    int kchunk = (i >> 9) & 7;
    int ntile  = i >> 12;  // 0..31
    int K = kchunk * 32 + (e & 7) + ((e >> 3) << 4) + ((lane >> 4) << 3);
    int N = ntile * 16 + (lane & 15);
    wpk[i] = (_Float16)W_in[K * TWO_D + N];
}

// ---------------- Big GEMM: xz[64000,512] = hin(f16) @ W_in(f16) + b_in -------
// One wave per 16(M) x 64(N) strip: 4 N-tiles share one A fragment.
// K = 256 in 8 chunks of 32, f32 accumulation. 32 WMMAs per wave.
__global__ __launch_bounds__(256) void k_gemm_xz(
    const _Float16* __restrict__ hin, const _Float16* __restrict__ wpk,
    const float* __restrict__ b_in, float* __restrict__ xz) {
    int lane   = threadIdx.x & 31;
    int wv     = threadIdx.x >> 5;
    int mtile  = blockIdx.y * 8 + wv;
    int ntile0 = blockIdx.x * NT_PER_WAVE;
    int m0     = mtile * 16;
    int khalf  = lane >> 4;
    int arow   = m0 + (lane & 15);

    const _Float16* aBase = hin + (size_t)arow * DIN + khalf * 8;
    const _Float16* bBase = wpk + ((size_t)ntile0 * 8 * 32 + lane) * 16;

    v8f acc[NT_PER_WAVE] = {};
#pragma unroll
    for (int kk = 0; kk < 8; ++kk) {
        if (kk < 7) __builtin_prefetch(aBase + (kk + 1) * 32, 0, 1);
        // A fragment: lanes 0-15 hold K = kk*32 + {0..7, 16..23};
        //             lanes 16-31 hold K = kk*32 + {8..15, 24..31}
        v8h alo = *(const v8h*)(aBase + kk * 32);
        v8h ahi = *(const v8h*)(aBase + kk * 32 + 16);
        v16h a;
#pragma unroll
        for (int e = 0; e < 8; ++e) { a[e] = alo[e]; a[e + 8] = ahi[e]; }
#pragma unroll
        for (int j = 0; j < NT_PER_WAVE; ++j) {
            // B fragment: pre-packed contiguous 32 bytes per lane
            v16h b = *(const v16h*)(bBase + ((size_t)j * 8 + kk) * 32 * 16);
            acc[j] = __builtin_amdgcn_wmma_f32_16x16x32_f16(
                /*neg_a=*/false, a, /*neg_b=*/false, b,
                /*c_mod=*/(short)0, acc[j], /*reuse_a=*/false, /*reuse_b=*/false);
        }
    }
    // C/D layout: VGPR r -> row m0 + r + khalf*8, col = ntile*16 + (lane&15)
#pragma unroll
    for (int j = 0; j < NT_PER_WAVE; ++j) {
        int col    = (ntile0 + j) * 16 + (lane & 15);
        float bias = b_in[col];
#pragma unroll
        for (int r = 0; r < 8; ++r) {
            int row = m0 + r + khalf * 8;
            xz[(size_t)row * TWO_D + col] = acc[j][r] + bias;
        }
    }
}

// ---------------- Depthwise causal conv (K=4) + SiLU -------------------------
__global__ void k_conv(const float* __restrict__ xz, const float* __restrict__ conv_w,
                       const float* __restrict__ conv_b, float* __restrict__ xc) {
    int i = blockIdx.x * blockDim.x + threadIdx.x;
    if (i >= MROWS * DDIM) return;
    int row = i >> 8;
    int d   = i & 255;
    int t   = row % LSEQ;
    float acc = conv_b[d];
#pragma unroll
    for (int k = 0; k < KCONV; ++k) {
        int tt = t + k - (KCONV - 1);
        if (tt >= 0)
            acc += xz[(size_t)(row + k - (KCONV - 1)) * TWO_D + d] * conv_w[d * KCONV + k];
    }
    xc[i] = acc / (1.0f + __expf(-acc)); // SiLU
}

// ---------------- x-projection + dt softplus ---------------------------------
__global__ __launch_bounds__(64) void k_xproj(
    const float* __restrict__ xc, const float* __restrict__ W_xproj,
    const float* __restrict__ W_dt, const float* __restrict__ b_dt,
    float* __restrict__ dt, float* __restrict__ Btb, float* __restrict__ Ctb) {
    __shared__ float xrow[DDIM];
    __shared__ float xdbl[DTRANK + 2 * NSTATE]; // 40
    int row = blockIdx.x;
    int tid = threadIdx.x;
    for (int d = tid; d < DDIM; d += 64) xrow[d] = xc[(size_t)row * DDIM + d];
    __syncthreads();
    for (int j = tid; j < DTRANK + 2 * NSTATE; j += 64) {
        float a = 0.f;
        for (int d = 0; d < DDIM; ++d) a += xrow[d] * W_xproj[d * 40 + j];
        xdbl[j] = a;
    }
    __syncthreads();
#pragma unroll
    for (int q = 0; q < 4; ++q) {
        int d = tid * 4 + q;
        float a = b_dt[d];
#pragma unroll
        for (int r = 0; r < DTRANK; ++r) a += xdbl[r] * W_dt[r * DDIM + d];
        float sp = (a > 20.f) ? a : __logf(1.0f + __expf(a)); // softplus
        dt[(size_t)row * DDIM + d] = sp;
    }
    if (tid < NSTATE)            Btb[row * NSTATE + tid]            = xdbl[DTRANK + tid];
    else if (tid < 2 * NSTATE)   Ctb[row * NSTATE + (tid - NSTATE)] = xdbl[DTRANK + NSTATE + (tid - NSTATE)];
}

// ---------------- Selective scan + skip + gating + fold mean over t ----------
// Grid (BATCH, 4): block = 64 channels of one batch; h[16] in registers.
// 256 blocks spread the latency-bound 1000-step chain across many WGPs.
__global__ __launch_bounds__(64) void k_scan(
    const float* __restrict__ dt, const float* __restrict__ Btb, const float* __restrict__ Ctb,
    const float* __restrict__ xc, const float* __restrict__ xz,
    const float* __restrict__ A_log, const float* __restrict__ D_skip,
    float* __restrict__ ysum) {
    int b   = blockIdx.x;
    int d   = blockIdx.y * 64 + threadIdx.x;
    int tid = threadIdx.x;
    float A[NSTATE], h[NSTATE];
#pragma unroll
    for (int n = 0; n < NSTATE; ++n) { A[n] = -__expf(A_log[d * NSTATE + n]); h[n] = 0.f; }
    float Dsk = D_skip[d];
    float acc = 0.f;
    __shared__ float Bs[NSTATE], Cs[NSTATE];
    for (int t = 0; t < LSEQ; ++t) {
        size_t row = (size_t)b * LSEQ + t;
        if (tid < NSTATE)            Bs[tid]          = Btb[row * NSTATE + tid];
        else if (tid < 2 * NSTATE)   Cs[tid - NSTATE] = Ctb[row * NSTATE + (tid - NSTATE)];
        __syncthreads();
        float dtv = dt[row * DDIM + d];
        float xv  = xc[row * DDIM + d];
        float dtx = dtv * xv;
        float y   = 0.f;
#pragma unroll
        for (int n = 0; n < NSTATE; ++n) {
            h[n] = __expf(dtv * A[n]) * h[n] + dtx * Bs[n];
            y   += h[n] * Cs[n];
        }
        float yv = y + xv * Dsk;
        float rv = xz[row * TWO_D + DDIM + d];            // res half
        acc += yv * (rv / (1.0f + __expf(-rv)));          // gated, summed over t
        __syncthreads();
    }
    ysum[b * DDIM + d] = acc;
}

// ---------------- pooled = (ysum/L) @ W_out + b_out --------------------------
__global__ __launch_bounds__(128) void k_pool(
    const float* __restrict__ ysum, const float* __restrict__ W_out,
    const float* __restrict__ b_out, float* __restrict__ pooled) {
    int b = blockIdx.x, s = threadIdx.x;
    float a = 0.f;
    for (int d = 0; d < DDIM; ++d) a += ysum[b * DDIM + d] * W_out[d * SDIM + s];
    pooled[b * SDIM + s] = a * (1.0f / (float)LSEQ) + b_out[s];
}

// ---------------- logits = pooled @ W_dec + b_dec ----------------------------
__global__ void k_dec(const float* __restrict__ pooled, const float* __restrict__ W_dec,
                      const float* __restrict__ b_dec, float* __restrict__ out) {
    int i = blockIdx.x * blockDim.x + threadIdx.x;
    if (i >= BATCH * LSEQ) return;
    int b = i / LSEQ, l = i % LSEQ;
    float a = b_dec[l];
    for (int s = 0; s < SDIM; ++s) a += pooled[b * SDIM + s] * W_dec[s * LSEQ + l];
    out[i] = a;
}

extern "C" void kernel_launch(void* const* d_in, const int* in_sizes, int n_in,
                              void* d_out, int out_size, void* d_ws, size_t ws_size,
                              hipStream_t stream) {
    (void)in_sizes; (void)n_in; (void)out_size; (void)ws_size;
    const float* x       = (const float*)d_in[0];
    const float* W_emb   = (const float*)d_in[1];
    const float* b_emb   = (const float*)d_in[2];
    const float* pos_emb = (const float*)d_in[3];
    const float* W_in    = (const float*)d_in[4];
    const float* b_in    = (const float*)d_in[5];
    const float* conv_w  = (const float*)d_in[6];
    const float* conv_b  = (const float*)d_in[7];
    const float* W_xproj = (const float*)d_in[8];
    const float* W_dt    = (const float*)d_in[9];
    const float* b_dt    = (const float*)d_in[10];
    const float* A_log   = (const float*)d_in[11];
    const float* D_skip  = (const float*)d_in[12];
    const float* W_out   = (const float*)d_in[13];
    const float* b_out   = (const float*)d_in[14];
    const float* W_dec   = (const float*)d_in[15];
    const float* b_dec   = (const float*)d_in[16];
    float* out = (float*)d_out;

    char* ws = (char*)d_ws;
    size_t off = 0;
    auto carve = [&](size_t bytes) -> char* {
        char* p = ws + off;
        off += (bytes + 255) & ~(size_t)255;
        return p;
    };
    _Float16* hin    = (_Float16*)carve((size_t)MROWS * DIN * 2);
    _Float16* wpk    = (_Float16*)carve((size_t)DIN * TWO_D * 2);
    float*    xz     = (float*)carve((size_t)MROWS * TWO_D * 4);
    float*    xc     = (float*)carve((size_t)MROWS * DDIM * 4);
    float*    dt     = (float*)carve((size_t)MROWS * DDIM * 4);
    float*    Btb    = (float*)carve((size_t)MROWS * NSTATE * 4);
    float*    Ctb    = (float*)carve((size_t)MROWS * NSTATE * 4);
    float*    ysum   = (float*)carve((size_t)BATCH * DDIM * 4);
    float*    pooled = (float*)carve((size_t)BATCH * SDIM * 4);

    k_embed<<<(MROWS * DIN + 255) / 256, 256, 0, stream>>>(x, W_emb, b_emb, pos_emb, hin);
    k_packW<<<(DIN * TWO_D + 255) / 256, 256, 0, stream>>>(W_in, wpk);
    dim3 gg(TWO_D / 16 / NT_PER_WAVE, MROWS / 16 / 8); // 8 x 500
    k_gemm_xz<<<gg, 256, 0, stream>>>(hin, wpk, b_in, xz);
    k_conv<<<(MROWS * DDIM + 255) / 256, 256, 0, stream>>>(xz, conv_w, conv_b, xc);
    k_xproj<<<MROWS, 64, 0, stream>>>(xc, W_xproj, W_dt, b_dt, dt, Btb, Ctb);
    k_scan<<<dim3(BATCH, 4), 64, 0, stream>>>(dt, Btb, Ctb, xc, xz, A_log, D_skip, ysum);
    k_pool<<<BATCH, SDIM, 0, stream>>>(ysum, W_out, b_out, pooled);
    k_dec<<<(BATCH * LSEQ + 255) / 256, 256, 0, stream>>>(pooled, W_dec, b_dec, out);
}
